// Attention_37091337568712
// MI455X (gfx1250) — matrix operand order repaired and checked by reading, not compiled, not run on hardware
//
#include <hip/hip_runtime.h>
#include <hip/hip_bf16.h>

typedef __attribute__((ext_vector_type(16))) _Float16 v16h;
typedef __attribute__((ext_vector_type(8)))  _Float16 v8h;
typedef __attribute__((ext_vector_type(8)))  float    v8f;
typedef __attribute__((ext_vector_type(4)))  unsigned int v4u;
typedef __attribute__((ext_vector_type(8)))  int      v8i;
typedef __attribute__((ext_vector_type(4)))  int      v4i;

#define HEADS    16
#define DIM_HEAD 64
#define SEQ      2048
#define DIM      1024
#define BATCH    2
#define MROWS    (BATCH * SEQ)      // 4096
#define SCALE    0.125f             // 64^-0.5

// ---------------------------------------------------------------------------
// Fragment helpers (v_wmma_f32_16x16x32_f16 register layouts per ISA 7.12.2)
// A 16x32: lane<16 -> row=lane, K0..7 in e0..7 and K16..23 in e8..15;
// lane>=16 -> K8..15 / K24..31. Two contiguous 16B chunks per lane.
// ---------------------------------------------------------------------------
static __device__ inline v16h make_a_frag(const _Float16* p0) {
    v8h lo = *(const v8h*)p0;
    v8h hi = *(const v8h*)(p0 + 16);
    v16h a;
#pragma unroll
    for (int i = 0; i < 8; ++i) { a[i] = lo[i]; a[i + 8] = hi[i]; }
    return a;
}

static __device__ inline v8f wmma_f16(v16h a, v16h b, v8f c) {
    return __builtin_amdgcn_wmma_f32_16x16x32_f16(false, a, false, b, (short)0, c,
                                                  false, false);
}

static __device__ inline unsigned short h_bits(_Float16 x) {
    return __builtin_bit_cast(unsigned short, x);
}

// LDS aperture maps generic addr[31:0] -> LDS byte offset (ISA 10.2)
static __device__ inline unsigned lds_off(const void* p) {
    return (unsigned)(unsigned long long)(size_t)p;
}

// ---------------------------------------------------------------------------
// Tensor Data Mover: 2D f16 tile load, Global -> LDS (ISA 08_async_tensor §8).
// D# group0: count=1 | lds_addr | global_addr[56:0] | type=2.
// D# group1: data_size=1(2B); tensor_dim0/1 == tile dims; dim0 stride given.
// Groups 2/3/4 zero (2D tensor). Issue from ONE wave; wait with TENSORcnt.
// This toolchain's builtin takes 6 args (extra int32x8 descriptor group).
// ---------------------------------------------------------------------------
static __device__ inline void tdm_load_2d_f16(unsigned lds_addr, const void* gptr,
                                              unsigned cols, unsigned rows,
                                              unsigned row_stride_elems) {
    unsigned long long ga = (unsigned long long)(size_t)gptr;
    v4u g0;
    g0[0] = 1u;                                        // count=1, user mode
    g0[1] = lds_addr;                                  // LDS byte address
    g0[2] = (unsigned)(ga & 0xFFFFFFFFu);              // global_addr[31:0]
    g0[3] = ((unsigned)(ga >> 32) & 0x01FFFFFFu)       // global_addr[56:32]
            | 0x80000000u;                             // type=2 ("image")
    v8i g1;
    g1[0] = (int)(1u << 16);                           // data_size=1 (2 bytes)
    g1[1] = (int)(cols << 16);                         // tensor_dim0[15:0]
    g1[2] = (int)((cols >> 16) | (rows << 16));        // dim0[31:16] | dim1[15:0]
    g1[3] = (int)((rows >> 16) | (cols << 16));        // dim1[31:16] | tile_dim0
    g1[4] = (int)(rows & 0xFFFFu);                     // tile_dim1 (tile_dim2=0)
    g1[5] = (int)row_stride_elems;                     // tensor_dim0_stride[31:0]
    g1[6] = 0;                                         // stride hi | dim1_stride lo
    g1[7] = 0;
    v4i g2; g2[0] = 0; g2[1] = 0; g2[2] = 0; g2[3] = 0;
    v4i g3; g3[0] = 0; g3[1] = 0; g3[2] = 0; g3[3] = 0;
    v8i g4;
#pragma unroll
    for (int i = 0; i < 8; ++i) g4[i] = 0;
    __builtin_amdgcn_tensor_load_to_lds(g0, g1, g2, g3, g4, 0);
}

// ---------------------------------------------------------------------------
// Kernel 1: w = x(f32)[4096,1024] @ Wqkv(f32)[1024,1024] -> f16 [4096,1024]
// Block tile 128x128, 8 waves (each 32x64 => 8 wmma / K-step), K-step 32.
// ---------------------------------------------------------------------------
__global__ __launch_bounds__(256) void gemm_qkv(const float* __restrict__ X,
                                                const float* __restrict__ W,
                                                _Float16* __restrict__ C) {
    __shared__ _Float16 As[128 * 32];    // [row][k]
    __shared__ _Float16 BsT[128 * 32];   // [n][k]

    const int tid  = threadIdx.x;
    const int lane = tid & 31;
    const int wv   = tid >> 5;
    const int wr   = wv & 3;             // row group: 32 rows
    const int wc   = wv >> 2;            // col group: 64 cols
    const int half = lane >> 4;
    const int l16  = lane & 15;
    const int rowBase = blockIdx.x * 128;
    const int colBase = blockIdx.y * 128;

    v8f acc[2][4];
#pragma unroll
    for (int u = 0; u < 2; ++u)
#pragma unroll
        for (int t = 0; t < 4; ++t)
#pragma unroll
            for (int v = 0; v < 8; ++v) acc[u][t][v] = 0.0f;

    for (int k0 = 0; k0 < DIM; k0 += 32) {
        __syncthreads();
        {   // A tile: 2 threads/row, 16 f32 each -> f16 contiguous
            const int r  = tid >> 1;
            const int c0 = (tid & 1) * 16;
            const float* src = X + (size_t)(rowBase + r) * DIM + k0 + c0;
            _Float16* dst = As + r * 32 + c0;
#pragma unroll
            for (int i = 0; i < 16; ++i) dst[i] = (_Float16)src[i];
        }
        {   // B tile transposed, packed b32 stores: thread -> (kpair, 8 cols)
            const int kp = tid >> 4;               // 0..15 -> k = 2kp, 2kp+1
            const int n0 = (tid & 15) * 8;
            const float* s0 = W + (size_t)(k0 + 2 * kp) * DIM + colBase + n0;
            const float* s1 = s0 + DIM;
            unsigned* dst = (unsigned*)BsT;
#pragma unroll
            for (int i = 0; i < 8; ++i) {
                unsigned lo = h_bits((_Float16)s0[i]);
                unsigned hi = h_bits((_Float16)s1[i]);
                dst[(n0 + i) * 16 + kp] = lo | (hi << 16);
            }
        }
        __syncthreads();

        v16h af[2];
#pragma unroll
        for (int u = 0; u < 2; ++u)
            af[u] = make_a_frag(As + (wr * 32 + u * 16 + l16) * 32 + half * 8);
#pragma unroll
        for (int t = 0; t < 4; ++t) {
            v16h bf = make_a_frag(BsT + (wc * 64 + t * 16 + l16) * 32 + half * 8);
#pragma unroll
            for (int u = 0; u < 2; ++u) acc[u][t] = wmma_f16(af[u], bf, acc[u][t]);
        }
    }

#pragma unroll
    for (int u = 0; u < 2; ++u)
#pragma unroll
        for (int t = 0; t < 4; ++t)
#pragma unroll
            for (int v = 0; v < 8; ++v) {
                const int r = rowBase + wr * 32 + u * 16 + v + 8 * half;
                const int c = colBase + wc * 64 + t * 16 + l16;
                C[(size_t)r * DIM + c] = (_Float16)acc[u][t][v];
            }
}

// ---------------------------------------------------------------------------
// Kernel 2: flash attention per (b,h): softmax(w w^T * SCALE), post-softmax
// column mask on the numerator only, times w. Block = (bh, 64 q rows), 4 waves.
// K/V tile staged by the Tensor Data Mover.
// ---------------------------------------------------------------------------
__global__ __launch_bounds__(128) void flash_attn(const _Float16* __restrict__ Wm,
                                                  const int* __restrict__ mask,
                                                  _Float16* __restrict__ O) {
    __shared__ _Float16 Ks [64 * 64];       // [key][feat]  (B operand for S)
    __shared__ _Float16 KsT[64 * 64];       // [feat][key]  (B operand for P*V)
    __shared__ _Float16 Ps [4 * 16 * 64];   // per-wave P staging (C->A relayout)

    const int tid  = threadIdx.x;
    const int lane = tid & 31;
    const int wv   = tid >> 5;              // 0..3
    const int half = lane >> 4;
    const int l16  = lane & 15;

    const int bh = blockIdx.x;              // 0..31
    const int b  = bh >> 4;
    const int h  = bh & 15;
    const int rowBlk = blockIdx.y;          // 0..31

    const _Float16* wb = Wm + (size_t)b * SEQ * DIM + h * DIM_HEAD; // row stride DIM
    const int* mrowb = mask + b * SEQ;
    const unsigned ksOff = lds_off(Ks);

    // Resident Q fragments: rows rowBlk*64 + wv*16 .. +15, 64 feats (2 K-chunks)
    const int qrow = rowBlk * 64 + wv * 16 + l16;
    v16h qf[2];
#pragma unroll
    for (int kc = 0; kc < 2; ++kc)
        qf[kc] = make_a_frag(wb + (size_t)qrow * DIM + kc * 32 + half * 8);

    float m_r[8], l_r[8];
    v8f   o[4];
#pragma unroll
    for (int v = 0; v < 8; ++v) { m_r[v] = -1.0e30f; l_r[v] = 0.0f; }
#pragma unroll
    for (int t = 0; t < 4; ++t)
#pragma unroll
        for (int v = 0; v < 8; ++v) o[t][v] = 0.0f;

    _Float16* ps = Ps + wv * 16 * 64;

    for (int cb = 0; cb < SEQ / 64; ++cb) {
        __syncthreads();                    // everyone done with previous Ks/KsT
        if (wv == 0) {                      // one wave drives the TDM
            tdm_load_2d_f16(ksOff, wb + (size_t)(cb * 64) * DIM, 64, 64, DIM);
            __builtin_amdgcn_s_wait_tensorcnt(0);
        }
        __syncthreads();
        {   // build KsT from Ks (LDS->LDS, packed b32 writes)
            const int jp = tid & 31;        // key pair: j = 2jp, 2jp+1
            const int f0 = (tid >> 5) * 16; // 16 feats per thread group
            const _Float16* r0 = Ks + (2 * jp) * 64 + f0;
            const _Float16* r1 = r0 + 64;
            unsigned* dstT = (unsigned*)KsT;
#pragma unroll
            for (int i = 0; i < 16; ++i) {
                unsigned lo = h_bits(r0[i]);
                unsigned hi = h_bits(r1[i]);
                dstT[(f0 + i) * 32 + jp] = lo | (hi << 16);
            }
        }
        __syncthreads();

        // ---- S = Q K^T * SCALE : four 16x16 tiles over 64 keys
        v8f s[4];
#pragma unroll
        for (int t = 0; t < 4; ++t) {
            v8f sa;
#pragma unroll
            for (int v = 0; v < 8; ++v) sa[v] = 0.0f;
#pragma unroll
            for (int kc = 0; kc < 2; ++kc) {
                v16h bf = make_a_frag(Ks + (t * 16 + l16) * 64 + kc * 32 + half * 8);
                sa = wmma_f16(qf[kc], bf, sa);
            }
#pragma unroll
            for (int v = 0; v < 8; ++v) s[t][v] = sa[v] * SCALE;
        }

        // ---- online softmax (row r = v + 8*half lives across one 16-lane half)
        float bm[8];
#pragma unroll
        for (int v = 0; v < 8; ++v)
            bm[v] = fmaxf(fmaxf(s[0][v], s[1][v]), fmaxf(s[2][v], s[3][v]));
#pragma unroll
        for (int off = 1; off < 16; off <<= 1)
#pragma unroll
            for (int v = 0; v < 8; ++v)
                bm[v] = fmaxf(bm[v], __shfl_xor(bm[v], off, 32));

        float corr[8];
#pragma unroll
        for (int v = 0; v < 8; ++v) {
            const float mn = fmaxf(m_r[v], bm[v]);
            corr[v] = __expf(m_r[v] - mn);
            m_r[v]  = mn;
        }
#pragma unroll
        for (int t = 0; t < 4; ++t)
#pragma unroll
            for (int v = 0; v < 8; ++v) o[t][v] *= corr[v];

        float rs[8];
#pragma unroll
        for (int v = 0; v < 8; ++v) rs[v] = 0.0f;
#pragma unroll
        for (int t = 0; t < 4; ++t)
#pragma unroll
            for (int v = 0; v < 8; ++v) {
                const float p = __expf(s[t][v] - m_r[v]);
                s[t][v] = p;                 // numerator candidate
                rs[v] += p;                  // denominator: UNMASKED sum
            }
#pragma unroll
        for (int off = 1; off < 16; off <<= 1)
#pragma unroll
            for (int v = 0; v < 8; ++v) rs[v] += __shfl_xor(rs[v], off, 32);
#pragma unroll
        for (int v = 0; v < 8; ++v) l_r[v] = l_r[v] * corr[v] + rs[v];

        // ---- post-softmax multiplicative mask on numerator, stage P to LDS
#pragma unroll
        for (int t = 0; t < 4; ++t) {
            const float mf = 1.0f - (float)mrowb[cb * 64 + t * 16 + l16];
#pragma unroll
            for (int v = 0; v < 8; ++v)
                ps[(v + 8 * half) * 64 + t * 16 + l16] = (_Float16)(s[t][v] * mf);
        }
        // in-wave cross-lane LDS exchange: stores visible before fragment loads
        asm volatile("s_wait_dscnt 0" ::: "memory");

        // ---- O += P V
        v16h pf[2];
#pragma unroll
        for (int kc = 0; kc < 2; ++kc)
            pf[kc] = make_a_frag(ps + l16 * 64 + kc * 32 + half * 8);
#pragma unroll
        for (int t = 0; t < 4; ++t)
#pragma unroll
            for (int kc = 0; kc < 2; ++kc) {
                v16h bf = make_a_frag(KsT + (t * 16 + l16) * 64 + kc * 32 + half * 8);
                o[t] = wmma_f16(pf[kc], bf, o[t]);
            }
    }

    // ---- finalize: divide by unmasked denominator, write f16 [b*n, h*64+d]
    float inv[8];
#pragma unroll
    for (int v = 0; v < 8; ++v) inv[v] = 1.0f / l_r[v];
#pragma unroll
    for (int t = 0; t < 4; ++t)
#pragma unroll
        for (int v = 0; v < 8; ++v) {
            const int r = rowBlk * 64 + wv * 16 + v + 8 * half;
            O[(size_t)(b * SEQ + r) * DIM + h * DIM_HEAD + t * 16 + l16] =
                (_Float16)(o[t][v] * inv[v]);
        }
}

// ---------------------------------------------------------------------------
// Kernel 3: out = attn(f16)[4096,1024] @ Wout(f32)[1024,1024] + bout -> f32
// Same 128x128 tiling; A tile (already f16) staged by the TDM while the other
// waves stage/convert the B tile.
// ---------------------------------------------------------------------------
__global__ __launch_bounds__(256) void gemm_out(const _Float16* __restrict__ A,
                                                const float* __restrict__ W,
                                                const float* __restrict__ bias,
                                                float* __restrict__ C) {
    __shared__ _Float16 As[128 * 32];
    __shared__ _Float16 BsT[128 * 32];

    const int tid  = threadIdx.x;
    const int lane = tid & 31;
    const int wv   = tid >> 5;
    const int wr   = wv & 3;
    const int wc   = wv >> 2;
    const int half = lane >> 4;
    const int l16  = lane & 15;
    const int rowBase = blockIdx.x * 128;
    const int colBase = blockIdx.y * 128;
    const unsigned asOff = lds_off(As);

    v8f acc[2][4];
#pragma unroll
    for (int u = 0; u < 2; ++u)
#pragma unroll
        for (int t = 0; t < 4; ++t)
#pragma unroll
            for (int v = 0; v < 8; ++v) acc[u][t][v] = 0.0f;

    for (int k0 = 0; k0 < DIM; k0 += 32) {
        __syncthreads();
        if (wv == 0)    // A tile: 128x32 f16, TDM async copy
            tdm_load_2d_f16(asOff, A + (size_t)rowBase * DIM + k0, 32, 128, DIM);
        {   // B tile transposed, f32 -> f16 packed b32 stores (overlaps TDM)
            const int kp = tid >> 4;
            const int n0 = (tid & 15) * 8;
            const float* s0 = W + (size_t)(k0 + 2 * kp) * DIM + colBase + n0;
            const float* s1 = s0 + DIM;
            unsigned* dst = (unsigned*)BsT;
#pragma unroll
            for (int i = 0; i < 8; ++i) {
                unsigned lo = h_bits((_Float16)s0[i]);
                unsigned hi = h_bits((_Float16)s1[i]);
                dst[(n0 + i) * 16 + kp] = lo | (hi << 16);
            }
        }
        if (wv == 0) __builtin_amdgcn_s_wait_tensorcnt(0);
        __syncthreads();

        v16h af[2];
#pragma unroll
        for (int u = 0; u < 2; ++u)
            af[u] = make_a_frag(As + (wr * 32 + u * 16 + l16) * 32 + half * 8);
#pragma unroll
        for (int t = 0; t < 4; ++t) {
            v16h bf = make_a_frag(BsT + (wc * 64 + t * 16 + l16) * 32 + half * 8);
#pragma unroll
            for (int u = 0; u < 2; ++u) acc[u][t] = wmma_f16(af[u], bf, acc[u][t]);
        }
    }

#pragma unroll
    for (int u = 0; u < 2; ++u)
#pragma unroll
        for (int t = 0; t < 4; ++t) {
            const int c = colBase + wc * 64 + t * 16 + l16;
            const float bv = bias[c];
#pragma unroll
            for (int v = 0; v < 8; ++v) {
                const int r = rowBase + wr * 32 + u * 16 + v + 8 * half;
                C[(size_t)r * DIM + c] = acc[u][t][v] + bv;
            }
        }
}

// ---------------------------------------------------------------------------
extern "C" void kernel_launch(void* const* d_in, const int* in_sizes, int n_in,
                              void* d_out, int out_size, void* d_ws, size_t ws_size,
                              hipStream_t stream) {
    (void)in_sizes; (void)n_in; (void)out_size; (void)ws_size;

    const float* x    = (const float*)d_in[0];
    const int*   mask = (const int*)  d_in[1];
    const float* Wqkv = (const float*)d_in[2];
    const float* Wout = (const float*)d_in[3];
    const float* bout = (const float*)d_in[4];
    float*       out  = (float*)d_out;

    _Float16* wbuf = (_Float16*)d_ws;                       // 4096*1024 f16
    _Float16* abuf = wbuf + (size_t)MROWS * DIM;            // 4096*1024 f16

    // 1) w = x @ Wqkv   (f16 result)
    gemm_qkv<<<dim3(MROWS / 128, DIM / 128), 256, 0, stream>>>(x, Wqkv, wbuf);

    // 2) flash attention per (b,h), 64-row q blocks
    flash_attn<<<dim3(BATCH * HEADS, SEQ / 64), 128, 0, stream>>>(wbuf, mask, abuf);

    // 3) out = attn @ Wout + bout   (f32 result)
    gemm_out<<<dim3(MROWS / 128, DIM / 128), 256, 0, stream>>>(abuf, Wout, bout, out);
}
